// GNN_7_78477642433200
// MI455X (gfx1250) — compile-verified, matrix operand order checked by reading, not run on hardware
//
#include <hip/hip_runtime.h>

// ---------------- Types for CDNA5 WMMA ----------------
typedef __attribute__((ext_vector_type(16))) __bf16 v16bf;
typedef __attribute__((ext_vector_type(8)))  float  v8f;

#define NN 10000
#define EE 160000
#define GG 256
#define CC 12

__device__ __forceinline__ unsigned short f2bf(float f) {
    unsigned int u = __float_as_uint(f);
    unsigned int r = u + 0x7FFFu + ((u >> 16) & 1u);   // round-to-nearest-even
    return (unsigned short)(r >> 16);
}

// ---------------- Elementwise helpers ----------------
__global__ void k_zero(float* __restrict__ p, long long n) {
    long long i = blockIdx.x * (long long)blockDim.x + threadIdx.x;
    if (i < n) p[i] = 0.f;
}

// msg = h[src]*edge_attr ; agg[dst] += msg   (edge-major, channel-minor for coalescing)
__global__ void k_scatter(const float* __restrict__ h,
                          const long long* __restrict__ src,
                          const long long* __restrict__ dst,
                          const float* __restrict__ ea,
                          float* __restrict__ agg, int ci) {
    long long t = blockIdx.x * (long long)blockDim.x + threadIdx.x;
    long long total = (long long)EE * ci;
    if (t >= total) return;
    int e = (int)(t / ci);
    int c = (int)(t % ci);
    int s = (int)src[e];
    int d = (int)dst[e];
    atomicAdd(&agg[(long long)d * ci + c], h[(long long)s * ci + c] * ea[e]);
}

// f32 [rows x cs] -> bf16 [rows x Kp], zero-padding K in [cs, Kp)
__global__ void k_tobf(const float* __restrict__ src, unsigned short* __restrict__ dst,
                       long long rows, int cs, int Kp) {
    long long t = blockIdx.x * (long long)blockDim.x + threadIdx.x;
    long long total = rows * (long long)Kp;
    if (t >= total) return;
    long long r = t / Kp;
    int k = (int)(t % Kp);
    dst[t] = (k < cs) ? f2bf(src[r * (long long)cs + k]) : (unsigned short)0;
}

// Pack W [K x M] (row-major, f32) into per-wave WMMA B-fragments (bf16 pairs in dwords),
// zero-padded to Kp rows.  Layout: out[((kt*mtiles+mt)<<8) + (lane<<3) + v]
__global__ void k_pack(const float* __restrict__ W, long long sW,
                       unsigned int* __restrict__ out, long long sO,
                       int K, int M, int Kp) {
    const float* Wh = W + blockIdx.y * sW;
    unsigned int* oh = out + blockIdx.y * sO;
    int mtiles = M >> 4;
    int total = (Kp >> 5) * mtiles * 256;
    int t = blockIdx.x * blockDim.x + threadIdx.x;
    if (t >= total) return;
    int v = t & 7, L = (t >> 3) & 31, tile = t >> 8;
    int mt = tile % mtiles, kt = tile / mtiles;
    int k0  = (kt << 5) + ((v & 4) << 2) + ((L >> 4) << 3) + ((v & 3) << 1);
    int col = (mt << 4) + (L & 15);
    unsigned int lo = (k0     < K) ? f2bf(Wh[(long long)k0       * M + col]) : 0u;
    unsigned int hi = (k0 + 1 < K) ? f2bf(Wh[(long long)(k0 + 1) * M + col]) : 0u;
    oh[t] = lo | (hi << 16);
}

// ---------------- WMMA GEMM kernels (1 wave == one 16x16 tile) ----------------
// GraphConv layer: out = relu( A1@B1 + A2@B2 + bias )   (dual accumulate into one v8f)
__global__ __launch_bounds__(32)
void k_gemm_dual(const unsigned short* __restrict__ A1, const unsigned int* __restrict__ B1,
                 const unsigned short* __restrict__ A2, const unsigned int* __restrict__ B2,
                 const float* __restrict__ bias, float* __restrict__ out,
                 int M, int Kp) {
    int lane = threadIdx.x;
    int mt = blockIdx.x, rt = blockIdx.y;
    int mtiles = M >> 4;
    int row  = (rt << 4) + (lane & 15);
    int koff = (lane >> 4) << 3;
    v8f acc = {};
    for (int kt = 0; kt < (Kp >> 5); ++kt) {
        union { unsigned int u[8]; v16bf v; } a1, a2, b1, b2;
        const unsigned short* r1 = A1 + (long long)row * Kp + (kt << 5);
        const unsigned short* r2 = A2 + (long long)row * Kp + (kt << 5);
        const unsigned int* bp1 = B1 + (((long long)kt * mtiles + mt) << 8) + (lane << 3);
        const unsigned int* bp2 = B2 + (((long long)kt * mtiles + mt) << 8) + (lane << 3);
#pragma unroll
        for (int v = 0; v < 8; ++v) {
            int k0 = ((v & 4) << 2) + koff + ((v & 3) << 1);
            a1.u[v] = *(const unsigned int*)(r1 + k0);
            a2.u[v] = *(const unsigned int*)(r2 + k0);
            b1.u[v] = bp1[v];
            b2.u[v] = bp2[v];
        }
        acc = __builtin_amdgcn_wmma_f32_16x16x32_bf16(false, a1.v, false, b1.v,
                                                      (short)0, acc, false, false);
        acc = __builtin_amdgcn_wmma_f32_16x16x32_bf16(false, a2.v, false, b2.v,
                                                      (short)0, acc, false, false);
    }
    int colb = (mt << 4) + (lane & 15);
    int rowb = (rt << 4) + ((lane >> 4) << 3);
    float bv = bias[colb];
#pragma unroll
    for (int r = 0; r < 8; ++r) {
        float val = acc[r] + bv;
        out[(long long)(rowb + r) * M + colb] = fmaxf(val, 0.f);
    }
}

// Batched (blockIdx.z = head) single-A GEMM: out = relu( A@B + bias )
__global__ __launch_bounds__(32)
void k_gemm(const unsigned short* __restrict__ A, long long sA,
            const unsigned int* __restrict__ B, long long sB,
            const float* __restrict__ bias, long long sBias,
            float* __restrict__ out, long long sOut,
            int M, int Kp) {
    int lane = threadIdx.x;
    int mt = blockIdx.x, rt = blockIdx.y, hd = blockIdx.z;
    const unsigned short* Ah = A + hd * sA;
    const unsigned int*   Bh = B + hd * sB;
    const float*          bh = bias + hd * sBias;
    float*                oh = out + hd * sOut;
    int mtiles = M >> 4;
    int row  = (rt << 4) + (lane & 15);
    int koff = (lane >> 4) << 3;
    v8f acc = {};
    for (int kt = 0; kt < (Kp >> 5); ++kt) {
        union { unsigned int u[8]; v16bf v; } a, b;
        const unsigned short* rp = Ah + (long long)row * Kp + (kt << 5);
        const unsigned int*   bp = Bh + (((long long)kt * mtiles + mt) << 8) + (lane << 3);
#pragma unroll
        for (int v = 0; v < 8; ++v) {
            int k0 = ((v & 4) << 2) + koff + ((v & 3) << 1);
            a.u[v] = *(const unsigned int*)(rp + k0);
            b.u[v] = bp[v];
        }
        acc = __builtin_amdgcn_wmma_f32_16x16x32_bf16(false, a.v, false, b.v,
                                                      (short)0, acc, false, false);
    }
    int colb = (mt << 4) + (lane & 15);
    int rowb = (rt << 4) + ((lane >> 4) << 3);
    float bv = bh[colb];
#pragma unroll
    for (int r = 0; r < 8; ++r) {
        float val = acc[r] + bv;
        oh[(long long)(rowb + r) * M + colb] = fmaxf(val, 0.f);
    }
}

// ---------------- Pooling + output head ----------------
__global__ void k_cnt(const long long* __restrict__ batch, float* __restrict__ cnt) {
    int n = blockIdx.x * blockDim.x + threadIdx.x;
    if (n < NN) atomicAdd(&cnt[(int)batch[n]], 1.f);
}
__global__ void k_poolacc(const float* __restrict__ h, const long long* __restrict__ batch,
                          float* __restrict__ pooled) {
    long long t = blockIdx.x * (long long)blockDim.x + threadIdx.x;
    if (t >= (long long)NN * 256) return;
    int n = (int)(t >> 8), c = (int)(t & 255);
    atomicAdd(&pooled[((long long)batch[n] << 8) + c], h[t]);
}
__global__ void k_pooldiv(float* __restrict__ pooled, const float* __restrict__ cnt) {
    int t = blockIdx.x * blockDim.x + threadIdx.x;
    if (t < GG * 256) pooled[t] /= fmaxf(cnt[t >> 8], 1.f);
}
// out[g, c] = h3[c][g] . outW[c] + outb[c]
__global__ void k_out(const float* __restrict__ h3, const float* __restrict__ oW,
                      const float* __restrict__ ob, float* __restrict__ out) {
    int t = blockIdx.x * blockDim.x + threadIdx.x;
    if (t >= GG * CC) return;
    int g = t / CC, c = t % CC;
    const float* hp = h3 + ((long long)c * GG + g) * 64;
    const float* wp = oW + c * 64;
    float s = ob[c];
#pragma unroll
    for (int i = 0; i < 64; ++i) s += hp[i] * wp[i];
    out[t] = s;
}

// ---------------- Launcher ----------------
static inline int blks(long long n, int b) { return (int)((n + b - 1) / b); }

extern "C" void kernel_launch(void* const* d_in, const int* in_sizes, int n_in,
                              void* d_out, int out_size, void* d_ws, size_t ws_size,
                              hipStream_t stream) {
    static const int CI[7] = {4, 32, 128, 256, 512, 512, 256};
    static const int CO[7] = {32, 128, 256, 512, 512, 256, 256};

    const float*     x   = (const float*)d_in[0];
    const long long* src = (const long long*)d_in[1];          // edge_index[0]
    const long long* dst = ((const long long*)d_in[1]) + EE;   // edge_index[1]
    const long long* bat = (const long long*)d_in[2];
    const float*     ea  = (const float*)d_in[3];

    const float *Wrel[7], *Brel[7], *Wroot[7];
    const float *mW1, *mb1, *mW2, *mb2, *mW3, *mb3, *oW, *ob;
    if (n_in >= 33) { // python lists expanded to individual inputs
        for (int l = 0; l < 7; ++l) {
            Wrel[l]  = (const float*)d_in[4 + l];
            Brel[l]  = (const float*)d_in[11 + l];
            Wroot[l] = (const float*)d_in[18 + l];
        }
        mW1 = (const float*)d_in[25]; mb1 = (const float*)d_in[26];
        mW2 = (const float*)d_in[27]; mb2 = (const float*)d_in[28];
        mW3 = (const float*)d_in[29]; mb3 = (const float*)d_in[30];
        oW  = (const float*)d_in[31]; ob  = (const float*)d_in[32];
    } else {          // python lists concatenated per entry
        const float* wr = (const float*)d_in[4];
        const float* br = (const float*)d_in[5];
        const float* wo = (const float*)d_in[6];
        long long offW = 0, offB = 0;
        for (int l = 0; l < 7; ++l) {
            Wrel[l]  = wr + offW;
            Wroot[l] = wo + offW;
            Brel[l]  = br + offB;
            offW += (long long)CI[l] * CO[l];
            offB += CO[l];
        }
        mW1 = (const float*)d_in[7];  mb1 = (const float*)d_in[8];
        mW2 = (const float*)d_in[9];  mb2 = (const float*)d_in[10];
        mW3 = (const float*)d_in[11]; mb3 = (const float*)d_in[12];
        oW  = (const float*)d_in[13]; ob  = (const float*)d_in[14];
    }

    // ---- workspace carve-up ----
    char* wsp = (char*)d_ws;
    auto alloc = [&](size_t bytes) -> void* {
        void* p = wsp;
        wsp += (bytes + 255) & ~(size_t)255;
        return p;
    };
    float*          agg  = (float*)alloc((size_t)NN * 512 * 4);
    float*          hA   = (float*)alloc((size_t)NN * 512 * 4);
    float*          hB   = (float*)alloc((size_t)NN * 512 * 4);
    unsigned short* abf  = (unsigned short*)alloc((size_t)NN * 512 * 2);
    unsigned short* hbf  = (unsigned short*)alloc((size_t)NN * 512 * 2);
    unsigned int*   bp1  = (unsigned int*)alloc((size_t)16 * 32 * 256 * 4);
    unsigned int*   bp2  = (unsigned int*)alloc((size_t)16 * 32 * 256 * 4);
    float*          pool = (float*)alloc((size_t)GG * 256 * 4);
    float*          cnt  = (float*)alloc((size_t)GG * 4);
    unsigned short* pbf  = (unsigned short*)alloc((size_t)GG * 256 * 2);
    float*          h1   = (float*)alloc((size_t)CC * GG * 256 * 4);
    unsigned short* h1bf = (unsigned short*)alloc((size_t)CC * GG * 256 * 2);
    float*          h2   = (float*)alloc((size_t)CC * GG * 128 * 4);
    unsigned short* h2bf = (unsigned short*)alloc((size_t)CC * GG * 128 * 2);
    float*          h3   = (float*)alloc((size_t)CC * GG * 64 * 4);
    unsigned int*   wp1  = (unsigned int*)alloc((size_t)CC * 8 * 16 * 256 * 4);
    unsigned int*   wp2  = (unsigned int*)alloc((size_t)CC * 8 * 8 * 256 * 4);
    unsigned int*   wp3  = (unsigned int*)alloc((size_t)CC * 4 * 4 * 256 * 4);

    // ---- 7 GraphConv layers ----
    const float* hcur = x;
    float* houts[2] = {hA, hB};
    for (int l = 0; l < 7; ++l) {
        int ci = CI[l], co = CO[l];
        int Kp = (ci + 31) & ~31;
        long long nagg = (long long)NN * ci;
        k_zero<<<blks(nagg, 256), 256, 0, stream>>>(agg, nagg);
        k_scatter<<<blks((long long)EE * ci, 256), 256, 0, stream>>>(hcur, src, dst, ea, agg, ci);
        k_tobf<<<blks((long long)NN * Kp, 256), 256, 0, stream>>>(agg, abf, NN, ci, Kp);
        k_tobf<<<blks((long long)NN * Kp, 256), 256, 0, stream>>>(hcur, hbf, NN, ci, Kp);
        int packN = (Kp >> 5) * (co >> 4) * 256;
        k_pack<<<dim3(blks(packN, 256), 1), 256, 0, stream>>>(Wrel[l], 0, bp1, 0, ci, co, Kp);
        k_pack<<<dim3(blks(packN, 256), 1), 256, 0, stream>>>(Wroot[l], 0, bp2, 0, ci, co, Kp);
        k_gemm_dual<<<dim3(co >> 4, NN >> 4), 32, 0, stream>>>(abf, bp1, hbf, bp2, Brel[l],
                                                               houts[l & 1], co, Kp);
        hcur = houts[l & 1];
    }

    // ---- global mean pool ----
    k_zero<<<blks(GG * 256, 256), 256, 0, stream>>>(pool, GG * 256);
    k_zero<<<blks(GG, 256), 256, 0, stream>>>(cnt, GG);
    k_cnt<<<blks(NN, 256), 256, 0, stream>>>(bat, cnt);
    k_poolacc<<<blks((long long)NN * 256, 256), 256, 0, stream>>>(hcur, bat, pool);
    k_pooldiv<<<blks(GG * 256, 256), 256, 0, stream>>>(pool, cnt);

    // ---- 12 MLP heads (batched over blockIdx.z) ----
    k_tobf<<<blks((long long)GG * 256, 256), 256, 0, stream>>>(pool, pbf, GG, 256, 256);

    int p1 = (256 >> 5) * (256 >> 4) * 256;   // W1: K=256, M=256
    k_pack<<<dim3(blks(p1, 256), CC), 256, 0, stream>>>(mW1, (long long)256 * 256, wp1, p1, 256, 256, 256);
    k_gemm<<<dim3(256 >> 4, GG >> 4, CC), 32, 0, stream>>>(pbf, 0, wp1, p1, mb1, 256,
                                                           h1, (long long)GG * 256, 256, 256);
    k_tobf<<<blks((long long)CC * GG * 256, 256), 256, 0, stream>>>(h1, h1bf, (long long)CC * GG, 256, 256);

    int p2 = (256 >> 5) * (128 >> 4) * 256;   // W2: K=256, M=128
    k_pack<<<dim3(blks(p2, 256), CC), 256, 0, stream>>>(mW2, (long long)256 * 128, wp2, p2, 256, 128, 256);
    k_gemm<<<dim3(128 >> 4, GG >> 4, CC), 32, 0, stream>>>(h1bf, (long long)GG * 256, wp2, p2, mb2, 128,
                                                           h2, (long long)GG * 128, 128, 256);
    k_tobf<<<blks((long long)CC * GG * 128, 256), 256, 0, stream>>>(h2, h2bf, (long long)CC * GG, 128, 128);

    int p3 = (128 >> 5) * (64 >> 4) * 256;    // W3: K=128, M=64
    k_pack<<<dim3(blks(p3, 256), CC), 256, 0, stream>>>(mW3, (long long)128 * 64, wp3, p3, 128, 64, 128);
    k_gemm<<<dim3(64 >> 4, GG >> 4, CC), 32, 0, stream>>>(h2bf, (long long)GG * 128, wp3, p3, mb3, 64,
                                                          h3, (long long)GG * 64, 64, 128);

    // ---- final 64 -> 1 projection, transposed output [G, C] ----
    k_out<<<blks(GG * CC, 256), 256, 0, stream>>>(h3, oW, ob, (float*)d_out);
}
